// Multi_head_attention_24773371363624
// MI455X (gfx1250) — matrix operand order repaired and checked by reading, not compiled, not run on hardware
//
#include <hip/hip_runtime.h>
#include <hip/hip_bf16.h>
#include <stdint.h>

// ---------------------------------------------------------------------------
// Problem constants
// ---------------------------------------------------------------------------
#define S_DIM 8192
#define E_DIM 4096
#define H_DIM 32
#define D_DIM 128
#define INV_SCALE (1.0f / 64.0f)   // 1/sqrt(4096)

typedef __attribute__((ext_vector_type(16))) __bf16          v16bf;
typedef __attribute__((ext_vector_type(8)))  float           v8f;
typedef __attribute__((ext_vector_type(8)))  unsigned short  ush8;

// fp32 -> bf16, round-to-nearest-even
__device__ __forceinline__ unsigned short f2bf(float f) {
    unsigned int u = __float_as_uint(f);
    u += 0x7FFFu + ((u >> 16) & 1u);
    return (unsigned short)(u >> 16);
}
__device__ __forceinline__ float bf2f(unsigned short h) {
    return __uint_as_float(((unsigned int)h) << 16);
}

// ---------------------------------------------------------------------------
// 1) elementwise fp32 -> bf16 conversion
// ---------------------------------------------------------------------------
__global__ __launch_bounds__(256) void cvt_f32_bf16(const float* __restrict__ src,
                                                    unsigned short* __restrict__ dst,
                                                    int n) {
    int i = blockIdx.x * 256 + threadIdx.x;
    if (i < n) dst[i] = f2bf(src[i]);
}

// Gather the 128 live rows of Wk (rows j = d*32) into bf16 [128 x 4096],
// plus the 128 live bias entries.
__global__ __launch_bounds__(256) void gather_wk(const float* __restrict__ Wk,
                                                 const float* __restrict__ bk,
                                                 unsigned short* __restrict__ wkb,
                                                 float* __restrict__ bkg) {
    int i = blockIdx.x * 256 + threadIdx.x;      // 0 .. 128*4096-1
    int d = i >> 12;                             // row (head-dim index)
    int e = i & 4095;                            // column
    wkb[i] = f2bf(Wk[(size_t)(d * H_DIM) * E_DIM + e]);
    if (i < D_DIM) bkg[i] = bk[i * H_DIM];
}

// ---------------------------------------------------------------------------
// 2) Tiled bf16 WMMA GEMM:  C[M,N] = A[M,K] * B[N,K]^T + bias[N]
//    Block tile 256(M) x 128(N), 8 waves (4x2), wave tile 64x64 -> 16 WMMAs
//    per K-step of 32.  Double-buffered LDS, one barrier per K-step.
// ---------------------------------------------------------------------------
union FragBF { v16bf v; ush8 h[2]; };

#define LDS_STRIDE 40   // 32 halves data + 8 pad; 80B rows, 16B-aligned

__global__ __launch_bounds__(256) void gemm_bf16_wmma(
        const unsigned short* __restrict__ A,   // [M x K] bf16
        const unsigned short* __restrict__ B,   // [N x K] bf16
        const float* __restrict__ bias,         // [N] fp32
        float* __restrict__ Cf,                 // fp32 output (or null)
        unsigned short* __restrict__ Cb,        // bf16 output (or null)
        int K, int ldc, int writeBF16) {
    __shared__ unsigned short As[2][256 * LDS_STRIDE];   // 2 x 20 KB
    __shared__ unsigned short Bs[2][128 * LDS_STRIDE];   // 2 x 10 KB

    const int tid   = threadIdx.x;
    const int lane  = tid & 31;
    const int wv    = tid >> 5;
    const int waveM = wv >> 1;          // 0..3  (64-row slabs of M)
    const int waveN = wv & 1;           // 0..1  (64-col slabs of N)
    const int bm    = blockIdx.y * 256;
    const int bn    = blockIdx.x * 128;
    const int l15   = lane & 15;
    const bool lo   = (lane < 16);

    // staging indices: A tile = 256 rows x 4 chunks, B tile = 128 rows x 4
    const int arow0 = tid >> 2, aq = (tid & 3) * 8;      // + i*64 rows
    const int brow0 = tid >> 2, bq = (tid & 3) * 8;      // + i*64 rows (i<2)

    v8f acc[4][4];
    const v8f vzero = {0.f, 0.f, 0.f, 0.f, 0.f, 0.f, 0.f, 0.f};
#pragma unroll
    for (int mi = 0; mi < 4; ++mi)
#pragma unroll
        for (int ni = 0; ni < 4; ++ni) acc[mi][ni] = vzero;

    const int nk = K >> 5;              // K-steps of 32

    // ---- prologue: stage K-slab 0 into buffer 0 ----
#pragma unroll
    for (int i = 0; i < 4; ++i)
        *(ush8*)&As[0][(arow0 + i * 64) * LDS_STRIDE + aq] =
            *(const ush8*)&A[(size_t)(bm + arow0 + i * 64) * K + aq];
#pragma unroll
    for (int i = 0; i < 2; ++i)
        *(ush8*)&Bs[0][(brow0 + i * 64) * LDS_STRIDE + bq] =
            *(const ush8*)&B[(size_t)(bn + brow0 + i * 64) * K + bq];
    __syncthreads();

    for (int kt = 0; kt < nk; ++kt) {
        const int p = kt & 1;
        const int knext = (kt + 1) << 5;

        // ---- issue global loads for K-slab kt+1 (overlaps WMMA below) ----
        ush8 ra[4], rb[2];
        if (kt + 1 < nk) {
#pragma unroll
            for (int i = 0; i < 4; ++i)
                ra[i] = *(const ush8*)&A[(size_t)(bm + arow0 + i * 64) * K + knext + aq];
#pragma unroll
            for (int i = 0; i < 2; ++i)
                rb[i] = *(const ush8*)&B[(size_t)(bn + brow0 + i * 64) * K + knext + bq];
        }
        if (kt + 2 < nk)   // hint slab kt+2 toward near caches
            __builtin_prefetch(&A[(size_t)(bm + (tid >> 1)) * K + knext + 32], 0, 3);

        // ---- fragments from LDS buffer p (documented 16-bit layouts) ----
        FragBF a[4], b[4];
#pragma unroll
        for (int mi = 0; mi < 4; ++mi) {
            int m = waveM * 64 + mi * 16 + l15;          // A row (M)
            // lanes 0-15: K 0-7 / 16-23 ; lanes 16-31: K 8-15 / 24-31
            a[mi].h[0] = *(const ush8*)&As[p][m * LDS_STRIDE + (lo ? 0 : 8)];
            a[mi].h[1] = *(const ush8*)&As[p][m * LDS_STRIDE + (lo ? 16 : 24)];
        }
#pragma unroll
        for (int ni = 0; ni < 4; ++ni) {
            int n = waveN * 64 + ni * 16 + l15;          // B column (N)
            // lanes 0-15: K 0-7 / 16-23 ; lanes 16-31: K 8-15 / 24-31
            b[ni].h[0] = *(const ush8*)&Bs[p][n * LDS_STRIDE + (lo ? 0 : 8)];
            b[ni].h[1] = *(const ush8*)&Bs[p][n * LDS_STRIDE + (lo ? 16 : 24)];
        }

        // ---- 16 WMMAs per wave per K-step ----
#pragma unroll
        for (int mi = 0; mi < 4; ++mi)
#pragma unroll
            for (int ni = 0; ni < 4; ++ni)
                acc[mi][ni] = __builtin_amdgcn_wmma_f32_16x16x32_bf16(
                    false, a[mi].v, false, b[ni].v,
                    (short)0, acc[mi][ni], false, false);

        // ---- drain staged regs into the other buffer ----
        if (kt + 1 < nk) {
            const int q = 1 - p;
#pragma unroll
            for (int i = 0; i < 4; ++i)
                *(ush8*)&As[q][(arow0 + i * 64) * LDS_STRIDE + aq] = ra[i];
#pragma unroll
            for (int i = 0; i < 2; ++i)
                *(ush8*)&Bs[q][(brow0 + i * 64) * LDS_STRIDE + bq] = rb[i];
        }
        __syncthreads();
    }

    // ---- epilogue: bias + store (C/D layout: lane<16 rows 0-7, lane>=16 rows 8-15)
#pragma unroll
    for (int mi = 0; mi < 4; ++mi) {
#pragma unroll
        for (int ni = 0; ni < 4; ++ni) {
            int nIdx = bn + waveN * 64 + ni * 16 + l15;
            float bb = bias[nIdx];
#pragma unroll
            for (int r = 0; r < 8; ++r) {
                int m = bm + waveM * 64 + mi * 16 + r + (lo ? 0 : 8);
                float val = acc[mi][ni][r] + bb;
                size_t off = (size_t)m * ldc + nIdx;
                if (writeBF16) Cb[off] = f2bf(val);
                else           Cf[off] = val;
            }
        }
    }
}

// ---------------------------------------------------------------------------
// 3) Fused align / mask / softmax / scale-V.  One wave32 per sequence row,
//    lane = head.  V (with bias) already lives in d_out; scaled in place.
// ---------------------------------------------------------------------------
__global__ __launch_bounds__(256) void attn_softmax_scale(
        const unsigned short* __restrict__ qb,   // [S x E] bf16
        const float* __restrict__ k0,            // [S x 128] fp32 (k[:, :, 0])
        const int* __restrict__ mask,            // [S x 32]
        float* __restrict__ out) {               // [S x E] fp32, holds V
    const int s    = blockIdx.x * 8 + (threadIdx.x >> 5);
    const int lane = threadIdx.x & 31;

    const unsigned short* qrow = qb + (size_t)s * E_DIM;
    const float*          krow = k0 + (size_t)s * D_DIM;

    float a = 0.f;
#pragma unroll 4
    for (int d = 0; d < D_DIM; ++d)
        a += krow[d] * bf2f(qrow[d * H_DIM + lane]);   // lanes read 64B slabs
    a *= INV_SCALE;
    if (mask[(size_t)s * H_DIM + lane] == 0) a = 1e-20f;

    // wave32 softmax across heads (lanes)
    float mx = a;
#pragma unroll
    for (int off = 16; off > 0; off >>= 1)
        mx = fmaxf(mx, __shfl_xor(mx, off, 32));
    float e = __expf(a - mx);
    float sum = e;
#pragma unroll
    for (int off = 16; off > 0; off >>= 1)
        sum += __shfl_xor(sum, off, 32);
    float w = e / sum;

    float* orow = out + (size_t)s * E_DIM;
#pragma unroll 4
    for (int d = 0; d < D_DIM; ++d)
        orow[d * H_DIM + lane] *= w;                   // coalesced 128B stores
}

// ---------------------------------------------------------------------------
// Workspace layout (bytes)
// ---------------------------------------------------------------------------
#define OFF_XB   ((size_t)0)            // x  bf16   : 8192*4096*2 = 64 MiB
#define OFF_WQB  ((size_t)67108864)     // Wq bf16   : 32 MiB
#define OFF_WVB  ((size_t)100663296)    // Wv bf16   : 32 MiB
#define OFF_WKB  ((size_t)134217728)    // Wk rows   : 128*4096*2 = 1 MiB
#define OFF_BKG  ((size_t)135266304)    // bk gather : 512 B
#define OFF_QB   ((size_t)135266816)    // q  bf16   : 64 MiB
#define OFF_K0   ((size_t)202375680)    // k0 fp32   : 8192*128*4 = 4 MiB
                                        // total ~197 MiB

extern "C" void kernel_launch(void* const* d_in, const int* in_sizes, int n_in,
                              void* d_out, int out_size, void* d_ws, size_t ws_size,
                              hipStream_t stream) {
    const float* x  = (const float*)d_in[0];
    const float* Wq = (const float*)d_in[1];
    const float* bq = (const float*)d_in[2];
    const float* Wk = (const float*)d_in[3];
    const float* bk = (const float*)d_in[4];
    const float* Wv = (const float*)d_in[5];
    const float* bv = (const float*)d_in[6];
    const int* mask = (const int*)d_in[7];
    float* out = (float*)d_out;

    char* ws = (char*)d_ws;
    unsigned short* xb  = (unsigned short*)(ws + OFF_XB);
    unsigned short* wqb = (unsigned short*)(ws + OFF_WQB);
    unsigned short* wvb = (unsigned short*)(ws + OFF_WVB);
    unsigned short* wkb = (unsigned short*)(ws + OFF_WKB);
    float*          bkg = (float*)(ws + OFF_BKG);
    unsigned short* qb  = (unsigned short*)(ws + OFF_QB);
    float*          k0  = (float*)(ws + OFF_K0);

    // 1) precision conversion (makes GEMM working set L2-resident: 96MB < 192MB)
    cvt_f32_bf16<<<(S_DIM * E_DIM) / 256, 256, 0, stream>>>(x,  xb,  S_DIM * E_DIM);
    cvt_f32_bf16<<<(E_DIM * E_DIM) / 256, 256, 0, stream>>>(Wq, wqb, E_DIM * E_DIM);
    cvt_f32_bf16<<<(E_DIM * E_DIM) / 256, 256, 0, stream>>>(Wv, wvb, E_DIM * E_DIM);
    gather_wk<<<(D_DIM * E_DIM) / 256, 256, 0, stream>>>(Wk, bk, wkb, bkg);

    // 2) the three projections via WMMA (block = 256 M x 128 N)
    dim3 gq(E_DIM / 128, S_DIM / 256);   // 32 x 32
    gemm_bf16_wmma<<<gq, 256, 0, stream>>>(xb, wqb, bq, nullptr, qb,
                                           E_DIM, E_DIM, 1);   // q -> bf16 ws
    dim3 gk(1, S_DIM / 256);             // only the 128 live K features
    gemm_bf16_wmma<<<gk, 256, 0, stream>>>(xb, wkb, bkg, k0, nullptr,
                                           E_DIM, D_DIM, 0);   // k0 -> fp32 ws
    dim3 gv(E_DIM / 128, S_DIM / 256);
    gemm_bf16_wmma<<<gv, 256, 0, stream>>>(xb, wvb, bv, out, nullptr,
                                           E_DIM, E_DIM, 0);   // v -> d_out

    // 3) fused align + mask + softmax + scale (in place on d_out)
    attn_softmax_scale<<<S_DIM / 8, 256, 0, stream>>>(qb, k0, mask, out);
}
